// VNAttention_1846835938132
// MI455X (gfx1250) — matrix operand order repaired and checked by reading, not compiled, not run on hardware
//
#include <hip/hip_runtime.h>
#include <math.h>

// ---------------------------------------------------------------------------
// VN-attention for MI455X (gfx1250, wave32).  Every GEMM runs on
// v_wmma_f32_16x16x32_f16; attention double-buffers K/V tiles into LDS with
// CDNA5 async global->LDS copies (ASYNCcnt).
//
// B=2, N=2048, C=256, coor=3, H=8, Dh=64, E = Dh*coor = 192, dim_inner = 512
// ---------------------------------------------------------------------------

typedef __attribute__((ext_vector_type(16))) _Float16 v16h;
typedef __attribute__((ext_vector_type(8)))  float    v8f;
typedef __attribute__((ext_vector_type(4)))  float    v4f;

union Frag {
  v16h h;
  v4f  f[2];
};

#define BATCH  2
#define NSEQ   2048
#define CDIM   256
#define HEADS  8
#define DHEAD  64
#define EDIM   192
#define DINNER 512
#define MROWS  (BATCH * NSEQ)          // 4096 (b,n) rows

// A-fragment (16x32 f16): lane(row=r16) holds K = {hl*8..+7} and {16+hl*8..+7}
__device__ __forceinline__ Frag load_a(const _Float16* base, int row0, int k0,
                                       int stride, int r16, int hl) {
  const _Float16* p = base + (size_t)(row0 + r16) * stride + k0 + hl * 8;
  Frag f;
  f.f[0] = *(const v4f*)(p);
  f.f[1] = *(const v4f*)(p + 16);
  return f;
}
// B-fragment (32x16 f16): lane(col=r16) holds K = {hl*8..+7} and {16+hl*8..+7}
__device__ __forceinline__ Frag load_b(const _Float16* base, int row0, int k0,
                                       int stride, int r16, int hl) {
  const _Float16* p = base + (size_t)(row0 + r16) * stride + k0 + hl * 8;
  Frag f;
  f.f[0] = *(const v4f*)(p);
  f.f[1] = *(const v4f*)(p + 16);
  return f;
}

// cross-lane reductions over the 16 lanes sharing a row-group
__device__ __forceinline__ float half_max16(float v) {
#pragma unroll
  for (int m = 1; m <= 8; m <<= 1) v = fmaxf(v, __shfl_xor(v, m, 32));
  return v;
}
__device__ __forceinline__ float half_sum16(float v) {
#pragma unroll
  for (int m = 1; m <= 8; m <<= 1) v += __shfl_xor(v, m, 32);
  return v;
}

// CDNA5 async global->LDS copy (ASYNCcnt); LDS byte address in VDST VGPR.
__device__ __forceinline__ void async_load_b128(unsigned lds_off, const void* g) {
  asm volatile("global_load_async_to_lds_b128 %0, %1, off"
               :: "v"(lds_off), "v"((unsigned long long)(size_t)g)
               : "memory");
}
__device__ __forceinline__ void wait_async0() {
  asm volatile("s_wait_asynccnt 0x0" ::: "memory");
}

// ---------------------------------------------------------------------------
// Kernel 0a: pack x (B,N,C,3) f32 -> xp[c][(b n)][i] f16
// ---------------------------------------------------------------------------
__global__ __launch_bounds__(256)
void vn_pack_x(const float* __restrict__ x, _Float16* __restrict__ xp) {
  const size_t idx = (size_t)blockIdx.x * 256 + threadIdx.x;
  const size_t total = (size_t)MROWS * CDIM * 3;
  if (idx >= total) return;
  const int c = (int)(idx % 3);
  const size_t rest = idx / 3;
  const int i = (int)(rest % CDIM);
  const size_t bn = rest / CDIM;
  xp[((size_t)c * MROWS + bn) * CDIM + i] = (_Float16)x[idx];
}

// ---------------------------------------------------------------------------
// Kernel 0b: weights -> f16 (qscale*log2e folded into Wq16)
// ---------------------------------------------------------------------------
__global__ __launch_bounds__(256)
void vn_pack_w(const float* __restrict__ Wq, const float* __restrict__ Wk,
               const float* __restrict__ Wv, const float* __restrict__ Wo,
               _Float16* __restrict__ wq16, _Float16* __restrict__ wk16,
               _Float16* __restrict__ wv16, _Float16* __restrict__ wo16,
               float qscale) {
  const int idx = blockIdx.x * 256 + threadIdx.x;     // 131072 each
  if (idx >= DINNER * CDIM) return;
  wq16[idx] = (_Float16)(Wq[idx] * qscale);
  wk16[idx] = (_Float16)Wk[idx];
  wv16[idx] = (_Float16)Wv[idx];
  wo16[idx] = (_Float16)Wo[idx];
}

// ---------------------------------------------------------------------------
// Kernel 1: WMMA GEMM for q/k/v.  Per wave: one 16-row tile x 32-col group,
// A-fragments (x rows) loaded once and reused across all three weights
// (48 WMMAs/wave).  Results scattered into attention layouts:
//   q,k: (bh, n, e=d*3+c) row-major;  v: transposed (bh, e, n).
// grid = 3 * (MROWS/64) * (DINNER/32), block = 128 (4 waves = 4 row tiles)
// ---------------------------------------------------------------------------
__global__ __launch_bounds__(128)
void vn_qkv_gemm(const _Float16* __restrict__ xp,
                 const _Float16* __restrict__ wq16,
                 const _Float16* __restrict__ wk16,
                 const _Float16* __restrict__ wv16,
                 _Float16* __restrict__ qo,
                 _Float16* __restrict__ ko,
                 _Float16* __restrict__ vo) {
  const int tid  = threadIdx.x;
  const int wave = tid >> 5;
  const int lane = tid & 31;
  const int hl   = lane >> 4;
  const int r16  = lane & 15;

  const int og = blockIdx.x % (DINNER / 32);
  const int rg = (blockIdx.x / (DINNER / 32)) % (MROWS / 64);
  const int c  = blockIdx.x / ((DINNER / 32) * (MROWS / 64));
  const int row0  = (rg * 4 + wave) * 16;
  const int obase = og * 32;

  const _Float16* xa = xp + (size_t)c * MROWS * CDIM;

  Frag af[8];
#pragma unroll
  for (int kc = 0; kc < 8; ++kc)
    af[kc] = load_a(xa, row0, kc * 32, CDIM, r16, hl);

  v8f accq[2], acck[2], accv[2];
  v8f zero = {};
#pragma unroll
  for (int ot = 0; ot < 2; ++ot) { accq[ot] = zero; acck[ot] = zero; accv[ot] = zero; }

#pragma unroll
  for (int kc = 0; kc < 8; ++kc) {
#pragma unroll
    for (int ot = 0; ot < 2; ++ot) {
      Frag bq = load_b(wq16, obase + ot * 16, kc * 32, CDIM, r16, hl);
      accq[ot] = __builtin_amdgcn_wmma_f32_16x16x32_f16(false, af[kc].h, false, bq.h,
                                                        (short)0, accq[ot], false, false);
      Frag bk = load_b(wk16, obase + ot * 16, kc * 32, CDIM, r16, hl);
      acck[ot] = __builtin_amdgcn_wmma_f32_16x16x32_f16(false, af[kc].h, false, bk.h,
                                                        (short)0, acck[ot], false, false);
      Frag bv = load_b(wv16, obase + ot * 16, kc * 32, CDIM, r16, hl);
      accv[ot] = __builtin_amdgcn_wmma_f32_16x16x32_f16(false, af[kc].h, false, bv.h,
                                                        (short)0, accv[ot], false, false);
    }
  }

  // scatter: C-layout lane holds col o = obase+ot*16+r16, rows r + 8*hl
#pragma unroll
  for (int ot = 0; ot < 2; ++ot) {
    const int o = obase + ot * 16 + r16;
    const int h = o >> 6, d = o & 63, e = d * 3 + c;
#pragma unroll
    for (int r = 0; r < 8; ++r) {
      const int bn = row0 + r + 8 * hl;
      const int b = bn >> 11, n = bn & (NSEQ - 1);
      const size_t qi = ((size_t)(b * HEADS + h) * NSEQ + n) * EDIM + e;
      qo[qi] = (_Float16)accq[ot][r];
      ko[qi] = (_Float16)acck[ot][r];
      vo[((size_t)(b * HEADS + h) * EDIM + e) * NSEQ + n] = (_Float16)accv[ot][r];
    }
  }
}

// ---------------------------------------------------------------------------
// Kernel 2: flash attention.  grid = BH * (NSEQ/64), block = 128 (4 waves),
// each wave owns 16 query rows; block double-buffers 32-key K/V tiles in LDS
// via async copies overlapped with WMMA compute.  Output written c-major:
// aop[c][(b n)][i = h*64+d] so the final projection is a clean GEMM.
// ---------------------------------------------------------------------------
__global__ __launch_bounds__(128)
void vn_attn_kernel(const _Float16* __restrict__ q,
                    const _Float16* __restrict__ k,
                    const _Float16* __restrict__ vtg,
                    _Float16* __restrict__ aop) {
  __shared__ __align__(16) _Float16 kbuf[2][32 * EDIM];   // 2 x 12 KB
  __shared__ __align__(16) _Float16 vbuf[2][EDIM * 32];   // 2 x 12 KB
  __shared__ __align__(16) _Float16 pstg[4][16 * 32];     // 4 KB P staging

  const int tid  = threadIdx.x;
  const int wave = tid >> 5;
  const int lane = tid & 31;
  const int hl   = lane >> 4;
  const int r16  = lane & 15;

  const int bh   = blockIdx.x >> 5;
  const int tile = blockIdx.x & 31;
  const int row0 = tile * 64 + wave * 16;
  const int b    = bh >> 3, h = bh & 7;

  const _Float16* qb = q   + (size_t)bh * NSEQ * EDIM;
  const _Float16* kb = k   + (size_t)bh * NSEQ * EDIM;
  const _Float16* vb = vtg + (size_t)bh * EDIM * NSEQ;

  Frag qa[6];
#pragma unroll
  for (int c = 0; c < 6; ++c) qa[c] = load_a(qb, row0, c * 32, EDIM, r16, hl);

  auto prefetch = [&](int buf, int jb) {
    const unsigned kl = (unsigned)(size_t)(&kbuf[buf][0]);
    const unsigned vl = (unsigned)(size_t)(&vbuf[buf][0]);
#pragma unroll
    for (int u = 0; u < 6; ++u) {
      const int cid = tid + u * 128;                 // 768 x 16B chunks each
      async_load_b128(kl + (unsigned)(cid * 16),
                      kb + (size_t)jb * EDIM + cid * 8);
      const int e = cid >> 2, part = cid & 3;
      async_load_b128(vl + (unsigned)(e * 64 + part * 16),
                      vb + (size_t)e * NSEQ + jb + part * 8);
    }
  };

  v8f zero = {};
  v8f oacc[12];
#pragma unroll
  for (int tt = 0; tt < 12; ++tt) oacc[tt] = zero;

  float m8[8], l8[8];
#pragma unroll
  for (int r = 0; r < 8; ++r) { m8[r] = -1e30f; l8[r] = 0.0f; }

  _Float16* pst = &pstg[wave][0];

  prefetch(0, 0);
  wait_async0();
  __syncthreads();

  for (int jb = 0; jb < NSEQ; jb += 32) {
    const int buf = (jb >> 5) & 1;
    if (jb + 32 < NSEQ) prefetch(buf ^ 1, jb + 32);

    const _Float16* kt = &kbuf[buf][0];
    const _Float16* vt = &vbuf[buf][0];

    // ---- S = Q * K^T ----
    v8f s0 = zero, s1 = zero;
#pragma unroll
    for (int c = 0; c < 6; ++c) {
      Frag kf0 = load_b(kt, 0, c * 32, EDIM, r16, hl);
      Frag kf1 = load_b(kt, 16, c * 32, EDIM, r16, hl);
      s0 = __builtin_amdgcn_wmma_f32_16x16x32_f16(false, qa[c].h, false, kf0.h,
                                                  (short)0, s0, false, false);
      s1 = __builtin_amdgcn_wmma_f32_16x16x32_f16(false, qa[c].h, false, kf1.h,
                                                  (short)0, s1, false, false);
    }

    // ---- online softmax ----
    float al[8];
#pragma unroll
    for (int r = 0; r < 8; ++r) {
      float mx = half_max16(fmaxf(s0[r], s1[r]));
      float mn = fmaxf(m8[r], mx);
      al[r] = exp2f(m8[r] - mn);
      m8[r] = mn;
      s0[r] = exp2f(s0[r] - mn);
      s1[r] = exp2f(s1[r] - mn);
      float rs = half_sum16(s0[r] + s1[r]);
      l8[r] = l8[r] * al[r] + rs;
    }
#pragma unroll
    for (int tt = 0; tt < 12; ++tt) {
#pragma unroll
      for (int r = 0; r < 8; ++r) oacc[tt][r] *= al[r];
    }

    // ---- stage P (f32 C-layout) -> LDS -> f16 A-layout fragment ----
#pragma unroll
    for (int r = 0; r < 8; ++r) {
      const int row = r + hl * 8;
      pst[row * 32 + r16]      = (_Float16)s0[r];
      pst[row * 32 + 16 + r16] = (_Float16)s1[r];
    }
    asm volatile("s_wait_dscnt 0" ::: "memory");
    Frag pa;
    pa.f[0] = *(const v4f*)(pst + r16 * 32 + hl * 8);
    pa.f[1] = *(const v4f*)(pst + r16 * 32 + 16 + hl * 8);

    // ---- O += P * V ----
#pragma unroll
    for (int tt = 0; tt < 12; ++tt) {
      Frag vf = load_b(vt, tt * 16, 0, 32, r16, hl);
      oacc[tt] = __builtin_amdgcn_wmma_f32_16x16x32_f16(false, pa.h, false, vf.h,
                                                        (short)0, oacc[tt], false, false);
    }

    wait_async0();
    __syncthreads();
  }

  // ---- normalize and scatter into aop[c][(b n)][h*64+d] ----
  float inv[8];
#pragma unroll
  for (int r = 0; r < 8; ++r) inv[r] = 1.0f / l8[r];
#pragma unroll
  for (int tt = 0; tt < 12; ++tt) {
    const int e = tt * 16 + r16;           // e = d*3 + c
    const int d = e / 3;
    const int c = e - d * 3;
    const int i = h * DHEAD + d;
#pragma unroll
    for (int r = 0; r < 8; ++r) {
      const int bn = b * NSEQ + row0 + r + 8 * hl;
      aop[((size_t)c * MROWS + bn) * DINNER + i] =
          (_Float16)(oacc[tt][r] * inv[r]);
    }
  }
}

// ---------------------------------------------------------------------------
// Kernel 3: WMMA GEMM out[bn,o,c] = sum_i aop[c][bn][i] * Wo[o][i], f32 out.
// grid = 3 * (MROWS/64) * (CDIM/32), block = 128 (4 waves = 4 row tiles)
// ---------------------------------------------------------------------------
__global__ __launch_bounds__(128)
void vn_out_gemm(const _Float16* __restrict__ aop,
                 const _Float16* __restrict__ wo16,
                 float* __restrict__ out) {
  const int tid  = threadIdx.x;
  const int wave = tid >> 5;
  const int lane = tid & 31;
  const int hl   = lane >> 4;
  const int r16  = lane & 15;

  const int og = blockIdx.x % (CDIM / 32);
  const int rg = (blockIdx.x / (CDIM / 32)) % (MROWS / 64);
  const int c  = blockIdx.x / ((CDIM / 32) * (MROWS / 64));
  const int row0  = (rg * 4 + wave) * 16;
  const int obase = og * 32;

  const _Float16* aa = aop + (size_t)c * MROWS * DINNER;

  v8f acc[2];
  v8f zero = {};
  acc[0] = zero; acc[1] = zero;

#pragma unroll
  for (int kc = 0; kc < 16; ++kc) {
    Frag af = load_a(aa, row0, kc * 32, DINNER, r16, hl);
#pragma unroll
    for (int ot = 0; ot < 2; ++ot) {
      Frag bf = load_b(wo16, obase + ot * 16, kc * 32, DINNER, r16, hl);
      acc[ot] = __builtin_amdgcn_wmma_f32_16x16x32_f16(false, af.h, false, bf.h,
                                                       (short)0, acc[ot], false, false);
    }
  }

#pragma unroll
  for (int ot = 0; ot < 2; ++ot) {
    const int o = obase + ot * 16 + r16;
#pragma unroll
    for (int r = 0; r < 8; ++r) {
      const int bn = row0 + r + 8 * hl;
      out[(size_t)bn * (CDIM * 3) + o * 3 + c] = acc[ot][r];
    }
  }
}

// ---------------------------------------------------------------------------
extern "C" void kernel_launch(void* const* d_in, const int* in_sizes, int n_in,
                              void* d_out, int out_size, void* d_ws, size_t ws_size,
                              hipStream_t stream) {
  (void)in_sizes; (void)n_in; (void)out_size; (void)ws_size;
  const float* x  = (const float*)d_in[0];
  const float* Wq = (const float*)d_in[1];
  const float* Wk = (const float*)d_in[2];
  const float* Wv = (const float*)d_in[3];
  const float* Wo = (const float*)d_in[4];
  float* out = (float*)d_out;

  // workspace layout (all f16 halves)
  const size_t XP  = (size_t)3 * MROWS * CDIM;          //  3,145,728
  const size_t WSZ = (size_t)DINNER * CDIM;             //    131,072 each
  const size_t QKV = (size_t)BATCH * HEADS * NSEQ * EDIM;  // 6,291,456 each
  const size_t AOP = (size_t)3 * MROWS * DINNER;        //  6,291,456

  _Float16* xp   = (_Float16*)d_ws;
  _Float16* wq16 = xp + XP;
  _Float16* wk16 = wq16 + WSZ;
  _Float16* wv16 = wk16 + WSZ;
  _Float16* wo16 = wv16 + WSZ;
  _Float16* q    = wo16 + WSZ;
  _Float16* k    = q + QKV;
  _Float16* v    = k + QKV;        // transposed (bh, e, n)
  _Float16* aop  = v + QKV;        // c-major (c, bn, i)

  // softmax scale 1/sqrt(192) folded into Wq along with log2(e) for exp2
  const float qscale = 1.44269504088896341f / sqrtf((float)EDIM);

  const int npack = (int)(((size_t)MROWS * CDIM * 3 + 255) / 256);
  vn_pack_x<<<npack, 256, 0, stream>>>(x, xp);
  vn_pack_w<<<(DINNER * CDIM + 255) / 256, 256, 0, stream>>>(
      Wq, Wk, Wv, Wo, wq16, wk16, wv16, wo16, qscale);
  vn_qkv_gemm<<<3 * (MROWS / 64) * (DINNER / 32), 128, 0, stream>>>(
      xp, wq16, wk16, wv16, q, k, v);
  vn_attn_kernel<<<BATCH * HEADS * (NSEQ / 64), 128, 0, stream>>>(q, k, v, aop);
  vn_out_gemm<<<3 * (MROWS / 64) * (CDIM / 32), 128, 0, stream>>>(aop, wo16, out);
}
